// Qwen3MoeSparseMoeBlock_12902081757849
// MI455X (gfx1250) — matrix operand order repaired and checked by reading, not compiled, not run on hardware
//
#include <hip/hip_runtime.h>

#define TB 4
#define TS 256
#define TD 2048
#define TE 32
#define TF 768
#define TK 8
#define TT (TB * TS)   // 1024 tokens

#define LDSA_STRIDE 40            // halves per row (80B, 16B-aligned, bank-spread)
#define LDSA_BUF    (64 * LDSA_STRIDE)

typedef __attribute__((ext_vector_type(16))) __bf16 v16bf;
typedef __attribute__((ext_vector_type(8)))  float  v8f;

#if defined(__HIP_DEVICE_COMPILE__)
#  if __has_builtin(__builtin_amdgcn_s_wait_asynccnt)
#    define WAIT_ASYNC(n) __builtin_amdgcn_s_wait_asynccnt(n)
#  else
#    define WAIT_ASYNC(n) asm volatile("s_wait_asynccnt " #n ::: "memory")
#  endif
#else
#  define WAIT_ASYNC(n)
#endif

// --- bf16 helpers -----------------------------------------------------------
__device__ __forceinline__ unsigned pack_bf16x2(float f0, float f1) {
    // v_perm_b32: truncated bf16(f0),bf16(f1) packed in one VALU op
    return __builtin_amdgcn_perm(__float_as_uint(f1), __float_as_uint(f0), 0x07060302u);
}
__device__ __forceinline__ unsigned short f2bfu(float f) {   // RNE, epilogue only
    unsigned u = __float_as_uint(f);
    return (unsigned short)((u + 0x7FFFu + ((u >> 16) & 1u)) >> 16);
}

// async copy of one 16B chunk global -> LDS (ASYNCcnt-tracked), inline asm:
// vdst = 32-bit LDS byte offset, vaddr = 64-bit global address (GV mode)
__device__ __forceinline__ void stage16B(const unsigned short* g, unsigned short* l) {
    unsigned loff = (unsigned)(size_t)(__attribute__((address_space(3))) unsigned short*)l;
    unsigned long long ga = (unsigned long long)(size_t)g;
    asm volatile("global_load_async_to_lds_b128 %0, %1, off"
                 :: "v"(loff), "v"(ga) : "memory");
}

// A fragment (16x32 bf16, contiguous k = 16*(lane/16) + s) from staged LDS tile
__device__ __forceinline__ v16bf lds_a_frag(const unsigned short* As, int row, int ka) {
    union { uint4 q[2]; v16bf v; } r;
    r.q[0] = *(const uint4*)(As + row * LDSA_STRIDE + ka);
    r.q[1] = *(const uint4*)(As + row * LDSA_STRIDE + 8 + ka);
    return r.v;
}

// B fragment (32x16 bf16, contiguous k = 16*(lane/16) + s): 16 contiguous fp32 -> 8 v_perm packs
__device__ __forceinline__ v16bf load_b_frag(const float* __restrict__ p) {
    union { unsigned u[8]; v16bf v; } r;
#pragma unroll
    for (int i = 0; i < 8; ++i) {
        float2 f = ((const float2*)p)[i];
        r.u[i] = pack_bf16x2(f.x, f.y);
    }
    return r.v;
}

// ---------------------------------------------------------------------------
// Router: one wave32 per token; lane == expert (E==32 == wave32).
// ---------------------------------------------------------------------------
__global__ __launch_bounds__(256) void moe_router(
    const float* __restrict__ x, const float* __restrict__ gw,
    int* __restrict__ topk_i, float* __restrict__ topk_w) {
    const int wid  = threadIdx.x >> 5;
    const int lane = threadIdx.x & 31;
    const int t = blockIdx.x * 8 + wid;
    if (t >= TT) return;

    const float4* x4 = (const float4*)(x  + (size_t)t    * TD);
    const float4* g4 = (const float4*)(gw + (size_t)lane * TD);
    float acc = 0.0f;
#pragma unroll 4
    for (int d = 0; d < TD / 4; ++d) {
        float4 a = x4[d], b = g4[d];
        acc = fmaf(a.x, b.x, fmaf(a.y, b.y, fmaf(a.z, b.z, fmaf(a.w, b.w, acc))));
    }

    float mx = acc;
    for (int off = 16; off > 0; off >>= 1) mx = fmaxf(mx, __shfl_xor(mx, off, 32));
    float p = __expf(acc - mx);
    float sum = p;
    for (int off = 16; off > 0; off >>= 1) sum += __shfl_xor(sum, off, 32);
    p /= sum;

    float wsum = 0.0f, sel_w = 0.0f;
    int sel_i = 0;
    float pv = p;
    for (int k = 0; k < TK; ++k) {
        float bv = pv; int bi = lane;
        for (int off = 16; off > 0; off >>= 1) {
            float ov = __shfl_xor(bv, off, 32);
            int   oi = __shfl_xor(bi, off, 32);
            if (ov > bv || (ov == bv && oi < bi)) { bv = ov; bi = oi; }
        }
        wsum += bv;
        if (lane == k)  { sel_i = bi; sel_w = bv; }
        if (lane == bi) pv = -1.0f;
    }
    if (lane < TK) {
        topk_i[t * TK + lane] = sel_i;
        topk_w[t * TK + lane] = sel_w / wsum;
    }
}

// ---------------------------------------------------------------------------
// Bookkeeping kernels
// ---------------------------------------------------------------------------
__global__ void moe_zero_counts(int* __restrict__ counts) {
    if (threadIdx.x < TE) counts[threadIdx.x] = 0;
}

__global__ void moe_zero_out(float4* __restrict__ out4, int n4) {
    int i = blockIdx.x * blockDim.x + threadIdx.x;
    if (i < n4) out4[i] = make_float4(0.f, 0.f, 0.f, 0.f);
}

__global__ void moe_xcvt(const float* __restrict__ x, unsigned short* __restrict__ xbf) {
    int id = blockIdx.x * blockDim.x + threadIdx.x;
    if (id >= TT * TD / 8) return;
    const float4* x4 = (const float4*)x;
    float4 a = x4[id * 2], b = x4[id * 2 + 1];
    uint4 o;
    o.x = pack_bf16x2(a.x, a.y); o.y = pack_bf16x2(a.z, a.w);
    o.z = pack_bf16x2(b.x, b.y); o.w = pack_bf16x2(b.z, b.w);
    ((uint4*)xbf)[id] = o;
}

__global__ void moe_assign(const int* __restrict__ topk_i, const float* __restrict__ topk_w,
                           int* __restrict__ counts, int* __restrict__ tok_of,
                           float* __restrict__ w_of) {
    int id = blockIdx.x * blockDim.x + threadIdx.x;
    if (id >= TT * TK) return;
    int e = topk_i[id];
    int pos = atomicAdd(&counts[e], 1);
    tok_of[e * TT + pos] = id / TK;
    w_of[e * TT + pos]   = topk_w[id];
}

__global__ void moe_scan(const int* __restrict__ counts, int* __restrict__ offsets) {
    if (threadIdx.x == 0) {
        int a = 0;
        for (int e = 0; e < TE; ++e) { offsets[e] = a; a += counts[e]; }
        offsets[TE] = a;
    }
}

// ---------------------------------------------------------------------------
// GEMM1: (expert, 64-token tile) x 256 F-cols: h = silu(x@wgT)*(x@wuT) -> bf16.
// A tile (64x32 bf16) staged once per K-step via async global->LDS copy,
// double-buffered (DMA for step s+1 overlaps WMMA of step s).
// Wave owns 4 M-tiles x 2 N-tiles for gate and up; each B fragment -> 4 WMMAs.
// ---------------------------------------------------------------------------
__global__ __launch_bounds__(256) void moe_gemm1(
    const unsigned short* __restrict__ xbf,
    const float* __restrict__ wg, const float* __restrict__ wu,
    const int* __restrict__ counts, const int* __restrict__ offsets,
    const int* __restrict__ tok_of, unsigned short* __restrict__ h_raw) {
    const int e    = blockIdx.x >> 4;
    const int tile = blockIdx.x & 15;
    const int cnt  = counts[e];
    const int base = tile * 64;
    if (base >= cnt) return;
    const int rows = min(64, cnt - base);

    __shared__ int s_tok[64];
    __shared__ unsigned short ldsA[2 * LDSA_BUF];
    if (threadIdx.x < 64)
        s_tok[threadIdx.x] = tok_of[e * TT + base + min((int)threadIdx.x, rows - 1)];
    __syncthreads();

    const int wid  = threadIdx.x >> 5;
    const int lane = threadIdx.x & 31;
    const int m    = lane & 15;
    const int hi   = lane >> 4;
    const int ka   = hi << 4;
    const int kb   = hi << 4;
    const int fbase = blockIdx.y * 256 + wid * 32;

    // per-thread 16B staging slot: row = tid/4, col-chunk = tid%4
    const int frow = threadIdx.x >> 2;
    const int fcc  = (threadIdx.x & 3) * 8;
    const unsigned short* gfill = xbf + (size_t)s_tok[frow] * TD + fcc;
    unsigned short* lfill = &ldsA[frow * LDSA_STRIDE + fcc];

    const size_t wb = (size_t)e * TF * TD;
    v8f accg[4][2] = {};
    v8f accu[4][2] = {};

    stage16B(gfill, lfill);                        // prefill buffer 0 (d0 = 0)

    const int NS = TD / 32;                        // 64 K-steps
    for (int s = 0; s < NS; ++s) {
        const int d0 = s * 32;
        if (s + 1 < NS) {
            stage16B(gfill + d0 + 32, lfill + ((s + 1) & 1) * LDSA_BUF);
            WAIT_ASYNC(1);                         // current buffer complete
        } else {
            WAIT_ASYNC(0);
        }
        __syncthreads();

        const unsigned short* As = &ldsA[(s & 1) * LDSA_BUF];
        v16bf a[4];
#pragma unroll
        for (int mt = 0; mt < 4; ++mt) a[mt] = lds_a_frag(As, mt * 16 + m, ka);

#pragma unroll
        for (int j = 0; j < 2; ++j) {
            const int fcol = fbase + j * 16 + m;
            const float* gp = wg + wb + (size_t)fcol * TD + d0 + kb;
            const float* up = wu + wb + (size_t)fcol * TD + d0 + kb;
            __builtin_prefetch(gp + 32, 0, 0);
            __builtin_prefetch(up + 32, 0, 0);
            v16bf bg = load_b_frag(gp);
            v16bf bu = load_b_frag(up);
#pragma unroll
            for (int mt = 0; mt < 4; ++mt) {
                accg[mt][j] = __builtin_amdgcn_wmma_f32_16x16x32_bf16(
                    false, a[mt], false, bg, (short)0, accg[mt][j], false, false);
                accu[mt][j] = __builtin_amdgcn_wmma_f32_16x16x32_bf16(
                    false, a[mt], false, bu, (short)0, accu[mt][j], false, false);
            }
        }
        __syncthreads();
    }

    const int hoff = offsets[e] + base;
#pragma unroll
    for (int mt = 0; mt < 4; ++mt) {
#pragma unroll
        for (int j = 0; j < 2; ++j) {
            const int col = fbase + j * 16 + m;
#pragma unroll
            for (int r = 0; r < 8; ++r) {
                const int lr = mt * 16 + r + (hi << 3);
                if (lr < rows) {
                    float g = accg[mt][j][r];
                    float hval = (g / (1.0f + __expf(-g))) * accu[mt][j][r];
                    h_raw[(size_t)(hoff + lr) * TF + col] = f2bfu(hval);
                }
            }
        }
    }
}

// ---------------------------------------------------------------------------
// GEMM2: (expert, 64-token tile) x 256 D-cols: y = h @ wd^T, weight-scaled,
// atomically combined. Same staged/double-buffered A path (h rows, bf16).
// ---------------------------------------------------------------------------
__global__ __launch_bounds__(256) void moe_gemm2(
    const unsigned short* __restrict__ h_raw, const float* __restrict__ wd,
    const int* __restrict__ counts, const int* __restrict__ offsets,
    const int* __restrict__ tok_of, const float* __restrict__ w_of,
    float* __restrict__ out) {
    const int e    = blockIdx.x >> 4;
    const int tile = blockIdx.x & 15;
    const int cnt  = counts[e];
    const int base = tile * 64;
    if (base >= cnt) return;
    const int rows = min(64, cnt - base);

    __shared__ int   s_tok[64];
    __shared__ float s_w[64];
    __shared__ unsigned short ldsA[2 * LDSA_BUF];
    if (threadIdx.x < 64) {
        bool v = (int)threadIdx.x < rows;
        s_tok[threadIdx.x] = v ? tok_of[e * TT + base + threadIdx.x] : 0;
        s_w[threadIdx.x]   = v ? w_of[e * TT + base + threadIdx.x]   : 0.0f;
    }
    __syncthreads();

    const int wid  = threadIdx.x >> 5;
    const int lane = threadIdx.x & 31;
    const int m    = lane & 15;
    const int hi   = lane >> 4;
    const int ka   = hi << 4;
    const int kb   = hi << 4;
    const int dbase = blockIdx.y * 256 + wid * 32;

    const int hoff = offsets[e] + base;
    const int frow = threadIdx.x >> 2;
    const int fcc  = (threadIdx.x & 3) * 8;
    const unsigned short* gfill =
        h_raw + (size_t)(hoff + min(frow, rows - 1)) * TF + fcc;
    unsigned short* lfill = &ldsA[frow * LDSA_STRIDE + fcc];

    const size_t wdb = (size_t)e * TD * TF;
    v8f acc[4][2] = {};

    stage16B(gfill, lfill);                        // prefill buffer 0 (f0 = 0)

    const int NS = TF / 32;                        // 24 K-steps
    for (int s = 0; s < NS; ++s) {
        const int f0 = s * 32;
        if (s + 1 < NS) {
            stage16B(gfill + f0 + 32, lfill + ((s + 1) & 1) * LDSA_BUF);
            WAIT_ASYNC(1);
        } else {
            WAIT_ASYNC(0);
        }
        __syncthreads();

        const unsigned short* As = &ldsA[(s & 1) * LDSA_BUF];
        v16bf a[4];
#pragma unroll
        for (int mt = 0; mt < 4; ++mt) a[mt] = lds_a_frag(As, mt * 16 + m, ka);

#pragma unroll
        for (int j = 0; j < 2; ++j) {
            const int dcol = dbase + j * 16 + m;
            const float* dp = wd + wdb + (size_t)dcol * TF + f0 + kb;
            __builtin_prefetch(dp + 32, 0, 0);
            v16bf b = load_b_frag(dp);
#pragma unroll
            for (int mt = 0; mt < 4; ++mt)
                acc[mt][j] = __builtin_amdgcn_wmma_f32_16x16x32_bf16(
                    false, a[mt], false, b, (short)0, acc[mt][j], false, false);
        }
        __syncthreads();
    }

#pragma unroll
    for (int mt = 0; mt < 4; ++mt) {
#pragma unroll
        for (int j = 0; j < 2; ++j) {
            const int dcol = dbase + j * 16 + m;
#pragma unroll
            for (int r = 0; r < 8; ++r) {
                const int lr = mt * 16 + r + (hi << 3);
                if (lr < rows) {
                    float val = acc[mt][j][r] * s_w[lr];
                    atomicAdd(&out[(size_t)s_tok[lr] * TD + dcol], val);
                }
            }
        }
    }
}

// ---------------------------------------------------------------------------
extern "C" void kernel_launch(void* const* d_in, const int* in_sizes, int n_in,
                              void* d_out, int out_size, void* d_ws, size_t ws_size,
                              hipStream_t stream) {
    (void)in_sizes; (void)n_in; (void)out_size; (void)ws_size;
    const float* x  = (const float*)d_in[0];   // [B,S,D] fp32
    const float* gw = (const float*)d_in[1];   // [E,D]
    const float* wg = (const float*)d_in[2];   // [E,F,D]
    const float* wu = (const float*)d_in[3];   // [E,F,D]
    const float* wd = (const float*)d_in[4];   // [E,D,F]
    float* out = (float*)d_out;                // [B,S,D] fp32

    char* p = (char*)d_ws;
    float* topk_w = (float*)p;            p += (size_t)TT * TK * sizeof(float);
    int*   topk_i = (int*)p;              p += (size_t)TT * TK * sizeof(int);
    int*   counts = (int*)p;              p += 64 * sizeof(int);
    int*   offs   = (int*)p;              p += 64 * sizeof(int);
    int*   tok_of = (int*)p;              p += (size_t)TE * TT * sizeof(int);
    float* w_of   = (float*)p;            p += (size_t)TE * TT * sizeof(float);
    unsigned short* xbf = (unsigned short*)p;     p += (size_t)TT * TD * sizeof(unsigned short);
    unsigned short* h_raw = (unsigned short*)p;   // [T*K, F] bf16 (compact)

    const int n4 = TT * TD / 4;
    moe_zero_counts<<<1, 64, 0, stream>>>(counts);
    moe_zero_out<<<(n4 + 255) / 256, 256, 0, stream>>>((float4*)out, n4);
    moe_xcvt<<<(TT * TD / 8 + 255) / 256, 256, 0, stream>>>(x, xbf);
    moe_router<<<TT / 8, 256, 0, stream>>>(x, gw, topk_i, topk_w);
    moe_assign<<<(TT * TK + 255) / 256, 256, 0, stream>>>(topk_i, topk_w, counts, tok_of, w_of);
    moe_scan<<<1, 32, 0, stream>>>(counts, offs);
    dim3 g1(TE * 16, TF / 256);
    moe_gemm1<<<g1, 256, 0, stream>>>(xbf, wg, wu, counts, offs, tok_of, h_raw);
    dim3 g2(TE * 16, TD / 256);
    moe_gemm2<<<g2, 256, 0, stream>>>(h_raw, wd, counts, offs, tok_of, w_of, out);
}
